// Image_Text_Attention_1065151889661
// MI455X (gfx1250) — matrix operand order repaired and checked by reading, not compiled, not run on hardware
//
#include <hip/hip_runtime.h>
#include <hip/hip_bf16.h>
#include <math.h>

#define B_  32
#define M_  196
#define DI_ 768
#define L_  512
#define K_  768

typedef __attribute__((ext_vector_type(16))) __bf16          v16bf;
typedef __attribute__((ext_vector_type(8)))  float           v8f;
typedef __attribute__((ext_vector_type(8)))  unsigned short  v8us;
typedef __attribute__((ext_vector_type(16))) unsigned short  v16us;

union ABfrag { v16us u; v16bf b; };

__device__ __forceinline__ unsigned short f32_to_bf16_rne(float f) {
  unsigned int u = __float_as_uint(f);
  u += 0x7FFFu + ((u >> 16) & 1u);   // round-to-nearest-even
  return (unsigned short)(u >> 16);
}

// ---------------------------------------------------------------------------
// Kernel 1: s[b,m] = tanh(image_emb . img_W + img_b) + 1
//           q[b,l] = tanh(seq_emb   . seq_W + seq_b)
// One wave (32 lanes) per 768-element dot product.
// ---------------------------------------------------------------------------
__global__ __launch_bounds__(256)
void proj_kernel(const float* __restrict__ img,
                 const float* __restrict__ seq,
                 const float* __restrict__ imgW,
                 const float* __restrict__ seqW,
                 const float* __restrict__ imgB,
                 const float* __restrict__ seqB,
                 float* __restrict__ s_out,
                 float* __restrict__ q_out) {
  const int PROWS = B_ * M_;          // 6272
  const int QROWS = B_ * L_;          // 16384
  int gwave = (int)((blockIdx.x * (unsigned)blockDim.x + threadIdx.x) >> 5);
  int lane  = threadIdx.x & 31;
  if (gwave >= PROWS + QROWS) return;

  const float* row; const float* w; float bias;
  bool isP = (gwave < PROWS);
  if (isP) { row = img + (size_t)gwave * DI_;           w = imgW; bias = imgB[0]; }
  else     { row = seq + (size_t)(gwave - PROWS) * K_;  w = seqW; bias = seqB[0]; }

  float acc = 0.f;
  for (int i = lane; i < DI_; i += 32) acc += row[i] * w[i];
  #pragma unroll
  for (int off = 16; off; off >>= 1) acc += __shfl_xor(acc, off, 32);

  if (lane == 0) {
    float t = tanhf(acc + bias);
    if (isP) s_out[gwave] = t + 1.0f;
    else     q_out[gwave - PROWS] = t;
  }
}

// ---------------------------------------------------------------------------
// Kernel 2: r[b,n] = sum_l q[b,l] * V[l,n]    (32x512x512 -> trivial)
// Block handles (b, 256 columns); q[b,:] staged in LDS.
// ---------------------------------------------------------------------------
__global__ __launch_bounds__(256)
void rv_kernel(const float* __restrict__ q,
               const float* __restrict__ V,
               float* __restrict__ r) {
  __shared__ float qs[L_];
  int b = blockIdx.x >> 1;
  int n = ((blockIdx.x & 1) << 8) + threadIdx.x;
  for (int i = threadIdx.x; i < L_; i += 256) qs[i] = q[(size_t)b * L_ + i];
  __syncthreads();
  float acc = 0.f;
  for (int l = 0; l < L_; ++l) acc += qs[l] * V[(size_t)l * L_ + n];
  r[(size_t)b * L_ + n] = acc;
}

// ---------------------------------------------------------------------------
// Kernel 3: seqT[b,k,l] = bf16(seq_emb[b,l,k])  (transpose so WMMA B-fragments
// are 16 contiguous K values per lane => one 32B load)
// ---------------------------------------------------------------------------
__global__ __launch_bounds__(256)
void tconv_kernel(const float* __restrict__ seq,
                  unsigned short* __restrict__ seqT) {
  __shared__ unsigned short tile[32][33];
  int b  = blockIdx.z;
  int l0 = blockIdx.y * 32;
  int k0 = blockIdx.x * 32;
  int tx = threadIdx.x, ty = threadIdx.y;       // 32 x 8

  const float* src = seq + ((size_t)b * L_ + l0) * K_ + k0;
  #pragma unroll
  for (int i = 0; i < 4; ++i) {
    int ll = ty + i * 8;
    tile[ll][tx] = f32_to_bf16_rne(src[(size_t)ll * K_ + tx]);
  }
  __syncthreads();
  unsigned short* dst = seqT + ((size_t)b * K_ + k0) * L_ + l0;
  #pragma unroll
  for (int i = 0; i < 4; ++i) {
    int kk = ty + i * 8;
    dst[(size_t)kk * L_ + tx] = tile[tx][kk];
  }
}

// ---------------------------------------------------------------------------
// Kernel 4: per (b, 16-row m-tile): masked softmax of s[m]*r[l], weights as
// bf16 A-tile in LDS, then [16x512]x[512x768] GEMM via v_wmma_f32_16x16x32_bf16.
// 256 threads = 8 waves; each wave owns 6 of the 48 N-tiles.
// ---------------------------------------------------------------------------
__global__ __launch_bounds__(256)
void fused_softmax_gemm(const float* __restrict__ s,
                        const float* __restrict__ r,
                        const int*   __restrict__ mask,
                        const unsigned short* __restrict__ seqT,
                        float* __restrict__ out) {
  __shared__ float r_s[L_];
  __shared__ float mk_s[L_];
  __shared__ float s_sh[16];
  __shared__ float red[16];
  __shared__ float partial[256];
  __shared__ float scaleRow[16];
  __shared__ __align__(32) unsigned short wA[16][520];   // padded: conflict-free

  const int mt  = blockIdx.x;
  const int b   = blockIdx.y;
  const int m0  = mt * 16;
  const int tid = threadIdx.x;
  const int lane = tid & 31;
  const int wv   = tid >> 5;

  // ---- load r, mask; masked max/min of r (for stable softmax) ----
  float lmax = -__builtin_inff(), lmin = __builtin_inff();
  for (int l = tid; l < L_; l += 256) {
    float rv = r[(size_t)b * L_ + l];
    int   mv = mask[(size_t)b * L_ + l];
    r_s[l]  = rv;
    mk_s[l] = mv ? 1.f : 0.f;
    if (mv) { lmax = fmaxf(lmax, rv); lmin = fminf(lmin, rv); }
  }
  if (tid < 16) {
    int gm = m0 + tid;
    s_sh[tid] = (gm < M_) ? s[(size_t)b * M_ + gm] : 0.f;
  }
  #pragma unroll
  for (int off = 16; off; off >>= 1) {
    lmax = fmaxf(lmax, __shfl_xor(lmax, off, 32));
    lmin = fminf(lmin, __shfl_xor(lmin, off, 32));
  }
  if (lane == 0) { red[wv] = lmax; red[8 + wv] = lmin; }
  __syncthreads();
  float rmax = red[0], rmin = red[8];
  #pragma unroll
  for (int i = 1; i < 8; ++i) {
    rmax = fmaxf(rmax, red[i]);
    rmin = fminf(rmin, red[8 + i]);
  }

  // ---- w[m,l] = exp(s_m*r_l - rowmax_m) (masked->0), row sums Z ----
  {
    const int mr = tid >> 4;     // 0..15 row
    const int lr = tid & 15;     // 16 threads per row, 32 l's each
    float a = s_sh[mr];
    float rowmax = (a >= 0.f) ? a * rmax : a * rmin;   // s>0 always, but general
    float zsum = 0.f;
    #pragma unroll 4
    for (int j = 0; j < 32; ++j) {
      int l = lr + j * 16;
      float val = (mk_s[l] != 0.f) ? __expf(a * r_s[l] - rowmax) : 0.f;
      wA[mr][l] = f32_to_bf16_rne(val);
      zsum += val;
    }
    partial[tid] = zsum;
  }
  __syncthreads();
  if (tid < 16) {
    float z = 0.f;
    #pragma unroll
    for (int j = 0; j < 16; ++j) z += partial[tid * 16 + j];
    scaleRow[tid] = (z > 0.f) ? (1.f / (z * sqrtf((float)K_))) : 0.f;
  }
  __syncthreads();

  // ---- WMMA GEMM: D[16,768] = wA[16,512] x seqT[b][512,768] ----
  const int half = lane >> 4;      // ISA A/B fragment half
  const int mrow = lane & 15;      // A row for this lane
  const int nn   = lane & 15;      // B/D column for this lane
  const unsigned short* bbase = seqT + (size_t)b * K_ * L_;
  const int koff = half * 16;      // B: lanes 16-31 hold K=16..31

  v8f acc[6] = {};
  for (int l0 = 0; l0 < L_; l0 += 32) {
    // A fragment (contiguous K, matches B): lanes<16 -> K {0..15},
    //                                       lanes>=16 -> K {16..31}
    v8us alo = *(const v8us*)&wA[mrow][l0 + half * 16];
    v8us ahi = *(const v8us*)&wA[mrow][l0 + half * 16 + 8];
    ABfrag af;
    af.u = __builtin_shufflevector(alo, ahi,
                                   0,1,2,3,4,5,6,7,8,9,10,11,12,13,14,15);
    #pragma unroll
    for (int t = 0; t < 6; ++t) {
      int n = (wv * 6 + t) * 16 + nn;
      const unsigned short* bp = bbase + (size_t)n * L_ + l0 + koff;
      if (l0 + 32 < L_) __builtin_prefetch(bp + 32, 0, 1);  // global_prefetch_b8
      ABfrag bf;
      bf.u = *(const v16us*)bp;    // 16 contiguous K values, 32B aligned
      acc[t] = __builtin_amdgcn_wmma_f32_16x16x32_bf16(
                   false, af.b, false, bf.b, (short)0, acc[t], false, false);
    }
  }

  // ---- store: D layout VGPR v -> M = v + half*8, N = lane&15 ----
  #pragma unroll
  for (int t = 0; t < 6; ++t) {
    int n0 = (wv * 6 + t) * 16;
    #pragma unroll
    for (int v = 0; v < 8; ++v) {
      int m  = v + half * 8;
      int gm = m0 + m;
      if (gm < M_)
        out[((size_t)b * M_ + gm) * K_ + n0 + nn] = acc[t][v] * scaleRow[m];
    }
  }
}

// ---------------------------------------------------------------------------
extern "C" void kernel_launch(void* const* d_in, const int* in_sizes, int n_in,
                              void* d_out, int out_size, void* d_ws, size_t ws_size,
                              hipStream_t stream) {
  const float* image_emb = (const float*)d_in[0];
  const float* seq_emb   = (const float*)d_in[1];
  const int*   mask      = (const int*)d_in[2];
  const float* img_W     = (const float*)d_in[3];
  const float* img_b     = (const float*)d_in[4];
  const float* seq_W     = (const float*)d_in[5];
  const float* seq_b     = (const float*)d_in[6];
  const float* V         = (const float*)d_in[7];
  float* out = (float*)d_out;

  char* ws = (char*)d_ws;
  float* s_buf = (float*)(ws);                        // 6272 f32  (25088 B)
  float* q_buf = (float*)(ws + 25088);                // 16384 f32 (65536 B)
  float* r_buf = (float*)(ws + 25088 + 65536);        // 16384 f32 (65536 B)
  unsigned short* seqT =
      (unsigned short*)(ws + 25088 + 65536 + 65536);  // B*K*L bf16 = 24 MB

  // 22656 rows total, 8 waves/block -> 2832 blocks
  proj_kernel<<<2832, 256, 0, stream>>>(image_emb, seq_emb, img_W, seq_W,
                                        img_b, seq_b, s_buf, q_buf);
  rv_kernel<<<64, 256, 0, stream>>>(q_buf, V, r_buf);
  tconv_kernel<<<dim3(K_ / 32, L_ / 32, B_), dim3(32, 8), 0, stream>>>(seq_emb, seqT);
  fused_softmax_gemm<<<dim3(13, B_), 256, 0, stream>>>(s_buf, r_buf, mask, seqT, out);
}
